// SelfAttention_63900523430484
// MI455X (gfx1250) — matrix operand order repaired and checked by reading, not compiled, not run on hardware
//
#include <hip/hip_runtime.h>

// ---------------------------------------------------------------------------
// Self-attention for gfx1250 (MI455X), wave32 + WMMA bf16 + async LDS staging.
//   Stage 1: proj_gemm_bf16 : Q = (X@WQ)/8, K = X@WK, Vt = (X@WV)^T  (bf16 out)
//   Stage 2: flash_attn     : per (b,h) flash attention; K/V blocks staged to
//            LDS via GLOBAL_LOAD_ASYNC_TO_LDS_B128 (double buffered, ASYNCcnt),
//            QK^T and P@V via v_wmma_f32_16x16x32_bf16.
// Workspace: 3 * 8192*1024 bf16 = 48 MB in d_ws.
// ---------------------------------------------------------------------------

typedef __attribute__((ext_vector_type(16))) __bf16        v16bf;
typedef __attribute__((ext_vector_type(8)))  float         v8f;
typedef __attribute__((ext_vector_type(4)))  unsigned int  u32x4;
typedef __attribute__((ext_vector_type(4)))  int           v4i;

#define INPUT_DIM 1024
#define HEADS     16
#define HEAD_DIM  64
#define OUT_DIM   1024
#define BATCH     4
#define SEQ       2048
#define MROWS     (BATCH * SEQ)   // 8192
#define NKB       (SEQ / 64)      // 32 key blocks

// ---- feature probes --------------------------------------------------------
#if defined(__has_builtin)
#if __has_builtin(__builtin_amdgcn_global_load_async_to_lds_b128) && \
    __has_builtin(__builtin_amdgcn_s_wait_asynccnt)
#define HAVE_ASYNC_LDS 1
#endif
#if __has_builtin(__builtin_amdgcn_cvt_pk_bf16_f32)
#define HAVE_PK_BF16 1
#endif
#endif

__device__ __forceinline__ unsigned short f32_bf16(float f) {
  unsigned int u = __builtin_bit_cast(unsigned int, f);
  u += 0x7FFFu + ((u >> 16) & 1u);          // round-to-nearest-even
  return (unsigned short)(u >> 16);
}

__device__ __forceinline__ unsigned int pack2_bf16(float a, float b) {
#ifdef HAVE_PK_BF16
  typedef __attribute__((ext_vector_type(2))) __bf16 v2bf;
  v2bf t = __builtin_amdgcn_cvt_pk_bf16_f32(a, b);
  return __builtin_bit_cast(unsigned int, t);
#else
  return (unsigned int)f32_bf16(a) | ((unsigned int)f32_bf16(b) << 16);
#endif
}

// 16-byte global -> LDS copy, async (ASYNCcnt) when available.
// Builtin signature (from hipcc diagnostic): param0 = AS1 (global) v4i*,
// param1 = AS3 (shared) v4i*, then imm offset, imm cpol.
__device__ __forceinline__ void stage16(const unsigned short* g, unsigned short* l) {
#ifdef HAVE_ASYNC_LDS
  __builtin_amdgcn_global_load_async_to_lds_b128(
      (__attribute__((address_space(1))) v4i*)g,
      (__attribute__((address_space(3))) v4i*)l, 0, 0);
#else
  *(u32x4*)l = *(const u32x4*)g;
#endif
}
__device__ __forceinline__ void stage_wait4() {
#ifdef HAVE_ASYNC_LDS
  __builtin_amdgcn_s_wait_asynccnt(4);
#endif
}
__device__ __forceinline__ void stage_wait0() {
#ifdef HAVE_ASYNC_LDS
  __builtin_amdgcn_s_wait_asynccnt(0);
#endif
}

// 16x32 (A) or 32x16 (B) bf16 fragment: 8 VGPRs, assembled via two b128 loads.
union Frag16 { u32x4 u[2]; v16bf v; };

// ---------------------------------------------------------------------------
// Projection GEMM: C[8192x1024] = (X[8192x1024] @ W[1024x1024]) * outScale.
// mode 0: out[((b*H+h)*S+s)*64+d]   (row-major per head;  Q and K)
// mode 1: out[((b*H+h)*64+d)*S+s]   (transposed per head; V^T)
// Block: 256 threads (8 waves), tile 128(M) x 64(N), K-step 32.
// ---------------------------------------------------------------------------
__global__ __launch_bounds__(256)
void proj_gemm_bf16(const float* __restrict__ X, const float* __restrict__ W,
                    unsigned short* __restrict__ Out, int mode, float outScale) {
  __shared__ __align__(16) unsigned short As[128][40];  // [m][k], pad 32->40
  __shared__ __align__(16) unsigned short Bs[64][40];   // [n][k] (W transposed)

  const int tid  = threadIdx.x;
  const int wave = tid >> 5;
  const int lane = tid & 31;
  const int l16  = lane & 15;
  const int lh   = lane >> 4;          // lane-half (0: lanes 0-15, 1: 16-31)
  const int wm   = wave & 3;
  const int wn   = wave >> 2;
  const int mBase = blockIdx.y * 128;
  const int nBase = blockIdx.x * 64;

  const v8f zero = {0.f, 0.f, 0.f, 0.f, 0.f, 0.f, 0.f, 0.f};
  v8f acc[2][2];
  acc[0][0] = zero; acc[0][1] = zero; acc[1][0] = zero; acc[1][1] = zero;

  for (int k0 = 0; k0 < INPUT_DIM; k0 += 32) {
    // A tile 128x32 fp32 -> bf16 LDS (1024 float4 / 256 threads)
#pragma unroll
    for (int it = 0; it < 4; ++it) {
      int v   = tid + it * 256;
      int row = v >> 3;               // 8 float4 per 32-wide row
      int c4  = v & 7;
      const float4 f = *(const float4*)(X + (size_t)(mBase + row) * INPUT_DIM + k0 + c4 * 4);
      *(unsigned int*)&As[row][c4 * 4]     = pack2_bf16(f.x, f.y);
      *(unsigned int*)&As[row][c4 * 4 + 2] = pack2_bf16(f.z, f.w);
    }
    // B tile: W[k0..k0+31][nBase..nBase+63] stored transposed Bs[n][k]
#pragma unroll
    for (int it = 0; it < 2; ++it) {
      int v  = tid + it * 256;        // 512 float4 total
      int kk = v >> 4;                // 16 float4 per 64-wide row
      int n4 = v & 15;
      const float4 f = *(const float4*)(W + (size_t)(k0 + kk) * OUT_DIM + nBase + n4 * 4);
      Bs[n4 * 4 + 0][kk] = f32_bf16(f.x);
      Bs[n4 * 4 + 1][kk] = f32_bf16(f.y);
      Bs[n4 * 4 + 2][kk] = f32_bf16(f.z);
      Bs[n4 * 4 + 3][kk] = f32_bf16(f.w);
    }
    __syncthreads();

    Frag16 af[2], bfm[2];
#pragma unroll
    for (int mt = 0; mt < 2; ++mt) {
      const unsigned short* r = &As[wm * 32 + mt * 16 + l16][0];
      af[mt].u[0] = *(const u32x4*)(r + lh * 8);        // K = 8*lh .. +7
      af[mt].u[1] = *(const u32x4*)(r + 16 + lh * 8);   // K = 16+8*lh .. +7
    }
#pragma unroll
    for (int nt = 0; nt < 2; ++nt) {
      const unsigned short* r = &Bs[wn * 32 + nt * 16 + l16][0];
      bfm[nt].u[0] = *(const u32x4*)(r + lh * 8);       // K = 8*lh .. +7
      bfm[nt].u[1] = *(const u32x4*)(r + 16 + lh * 8);  // K = 16+8*lh .. +7
    }
#pragma unroll
    for (int mt = 0; mt < 2; ++mt)
#pragma unroll
      for (int nt = 0; nt < 2; ++nt)
        acc[mt][nt] = __builtin_amdgcn_wmma_f32_16x16x32_bf16(
            false, af[mt].v, false, bfm[nt].v, (short)0, acc[mt][nt], false, false);
    __syncthreads();
  }

  // Epilogue: C element i of lane L = row (L/16)*8+i, col L%16 of its tile.
#pragma unroll
  for (int mt = 0; mt < 2; ++mt) {
#pragma unroll
    for (int nt = 0; nt < 2; ++nt) {
      const int M0   = mBase + wm * 32 + mt * 16 + lh * 8;
      const int Ncol = nBase + wn * 32 + nt * 16 + l16;
      const int h = Ncol >> 6, d = Ncol & 63;
#pragma unroll
      for (int e = 0; e < 8; ++e) {
        const int M = M0 + e;
        const int b = M >> 11;          // / SEQ
        const int s = M & (SEQ - 1);
        size_t addr = (mode == 0)
            ? ((((size_t)b * HEADS + h) * SEQ + s) * HEAD_DIM + d)
            : ((((size_t)b * HEADS + h) * HEAD_DIM + d) * SEQ + s);
        Out[addr] = f32_bf16(acc[mt][nt][e] * outScale);
      }
    }
  }
}

// ---------------------------------------------------------------------------
// Flash attention. Grid: (SEQ/128, B*H), block 256 (8 waves).
// Each wave owns 16 query rows. 64-key K/V blocks are staged into LDS once per
// workgroup (async, double buffered) and shared by all 8 waves.
//   Q  [bh][s][d]  (A fragments, contraction over d; pre-scaled by 1/sqrt(D))
//   K  [bh][s][d]  -> LDS Ks[key][d]   (B frags for QK^T: [n=key][k=d])
//   Vt [bh][d][s]  -> LDS Vs[d][key]   (B frags for P@V : [n=d][k=key])
// ---------------------------------------------------------------------------
__global__ __launch_bounds__(256)
void flash_attn(const unsigned short* __restrict__ Q,
                const unsigned short* __restrict__ K,
                const unsigned short* __restrict__ Vt,
                float* __restrict__ Out) {
  __shared__ __align__(16) unsigned short Ks[2][64][72];  // [buf][key][d]
  __shared__ __align__(16) unsigned short Vs[2][64][72];  // [buf][d][key]
  __shared__ __align__(16) unsigned short Pl[8][16][72];  // per-wave P scratch

  const int tid  = threadIdx.x;
  const int wave = tid >> 5;
  const int lane = tid & 31;
  const int l16  = lane & 15;
  const int lh   = lane >> 4;
  const int bh   = blockIdx.y;
  const int b    = bh >> 4;            // / HEADS
  const int h    = bh & (HEADS - 1);
  const int q0   = (blockIdx.x * 8 + wave) * 16;

  const unsigned short* Qb = Q  + (size_t)bh * SEQ * HEAD_DIM;
  const unsigned short* Kb = K  + (size_t)bh * SEQ * HEAD_DIM;
  const unsigned short* Vb = Vt + (size_t)bh * HEAD_DIM * SEQ;

  // Stage one 64-key block (8 KB K + 8 KB V): 4 x 16B async copies per thread.
  auto stage = [&](int j, int buf) {
    const int kb0 = j * 64;
#pragma unroll
    for (int it = 0; it < 2; ++it) {
      int id = tid + it * 256;        // 512 16B chunks each for K and V
      int r  = id >> 3;
      int c  = (id & 7) * 8;
      stage16(Kb + (size_t)(kb0 + r) * HEAD_DIM + c, &Ks[buf][r][c]);
      stage16(Vb + (size_t)r * SEQ + kb0 + c,        &Vs[buf][r][c]);
    }
  };

  // Q fragments for this wave's 16 rows (kept in registers for whole loop).
  Frag16 qa[2];
#pragma unroll
  for (int kc = 0; kc < 2; ++kc) {
    const unsigned short* r = Qb + (size_t)(q0 + l16) * HEAD_DIM + kc * 32;
    qa[kc].u[0] = *(const u32x4*)(r + lh * 8);
    qa[kc].u[1] = *(const u32x4*)(r + 16 + lh * 8);
  }

  const v8f zero = {0.f, 0.f, 0.f, 0.f, 0.f, 0.f, 0.f, 0.f};
  v8f acc[4];
  acc[0] = zero; acc[1] = zero; acc[2] = zero; acc[3] = zero;
  float mrow[8], lrow[8];
#pragma unroll
  for (int e = 0; e < 8; ++e) { mrow[e] = -1e30f; lrow[e] = 0.f; }

  stage(0, 0);                         // preload block 0

  for (int j = 0; j < NKB; ++j) {
    const int buf = j & 1;
    if (j + 1 < NKB) { stage(j + 1, buf ^ 1); stage_wait4(); }  // pipeline
    else             { stage_wait0(); }
    __syncthreads();                   // block j resident in LDS for all waves

    // ---- scores: S = Q K^T (Q pre-scaled by 1/8), 16x64 as 4 C-frags ----
    v8f sc[4];
#pragma unroll
    for (int nt = 0; nt < 4; ++nt) {
      Frag16 kf0, kf1;
      const unsigned short* r = &Ks[buf][nt * 16 + l16][0];
      kf0.u[0] = *(const u32x4*)(r + lh * 8);
      kf0.u[1] = *(const u32x4*)(r + 16 + lh * 8);
      kf1.u[0] = *(const u32x4*)(r + 32 + lh * 8);
      kf1.u[1] = *(const u32x4*)(r + 48 + lh * 8);
      v8f t = __builtin_amdgcn_wmma_f32_16x16x32_bf16(
          false, qa[0].v, false, kf0.v, (short)0, zero, false, false);
      sc[nt] = __builtin_amdgcn_wmma_f32_16x16x32_bf16(
          false, qa[1].v, false, kf1.v, (short)0, t, false, false);
    }

    // ---- online softmax (element e of lane = row lh*8+e, col l16) ----
    float rmax[8];
#pragma unroll
    for (int e = 0; e < 8; ++e) {
      float v = fmaxf(fmaxf(sc[0][e], sc[1][e]), fmaxf(sc[2][e], sc[3][e]));
      v = fmaxf(v, __shfl_xor(v, 1));
      v = fmaxf(v, __shfl_xor(v, 2));
      v = fmaxf(v, __shfl_xor(v, 4));
      v = fmaxf(v, __shfl_xor(v, 8));
      rmax[e] = v;
    }
#pragma unroll
    for (int e = 0; e < 8; ++e) {
      float mnew  = fmaxf(mrow[e], rmax[e]);
      float scale = __expf(mrow[e] - mnew);
      float rs = 0.f;
#pragma unroll
      for (int nt = 0; nt < 4; ++nt) {
        float p = __expf(sc[nt][e] - mnew);
        sc[nt][e] = p;
        rs += p;
      }
      rs += __shfl_xor(rs, 1);
      rs += __shfl_xor(rs, 2);
      rs += __shfl_xor(rs, 4);
      rs += __shfl_xor(rs, 8);
      lrow[e] = lrow[e] * scale + rs;
      mrow[e] = mnew;
#pragma unroll
      for (int dt = 0; dt < 4; ++dt) acc[dt][e] *= scale;
    }

    // ---- re-shape P: C-layout -> A-layout via per-wave LDS ----
#pragma unroll
    for (int nt = 0; nt < 4; ++nt)
#pragma unroll
      for (int e = 0; e < 8; ++e)
        Pl[wave][lh * 8 + e][nt * 16 + l16] = f32_bf16(sc[nt][e]);
    __syncthreads();

    Frag16 pa[2];
#pragma unroll
    for (int kc = 0; kc < 2; ++kc) {
      const unsigned short* r = &Pl[wave][l16][kc * 32];
      pa[kc].u[0] = *(const u32x4*)(r + lh * 8);
      pa[kc].u[1] = *(const u32x4*)(r + 16 + lh * 8);
    }

    // ---- acc += P @ V (contraction over 64 keys) ----
#pragma unroll
    for (int dt = 0; dt < 4; ++dt) {
      Frag16 vf0, vf1;
      const unsigned short* r = &Vs[buf][dt * 16 + l16][0];
      vf0.u[0] = *(const u32x4*)(r + lh * 8);
      vf0.u[1] = *(const u32x4*)(r + 16 + lh * 8);
      vf1.u[0] = *(const u32x4*)(r + 32 + lh * 8);
      vf1.u[1] = *(const u32x4*)(r + 48 + lh * 8);
      acc[dt] = __builtin_amdgcn_wmma_f32_16x16x32_bf16(
          false, pa[0].v, false, vf0.v, (short)0, acc[dt], false, false);
      acc[dt] = __builtin_amdgcn_wmma_f32_16x16x32_bf16(
          false, pa[1].v, false, vf1.v, (short)0, acc[dt], false, false);
    }
    __syncthreads();                   // reads done before next stage overwrites
  }

  // ---- epilogue: out[b][s][h*64+d] = acc / l ----
#pragma unroll
  for (int dt = 0; dt < 4; ++dt) {
    const int d = dt * 16 + l16;
#pragma unroll
    for (int e = 0; e < 8; ++e) {
      const int s = q0 + lh * 8 + e;
      Out[((size_t)b * SEQ + s) * OUT_DIM + h * HEAD_DIM + d] = acc[dt][e] / lrow[e];
    }
  }
}

// ---------------------------------------------------------------------------
extern "C" void kernel_launch(void* const* d_in, const int* in_sizes, int n_in,
                              void* d_out, int out_size, void* d_ws, size_t ws_size,
                              hipStream_t stream) {
  const float* query = (const float*)d_in[0];
  const float* key   = (const float*)d_in[1];
  const float* value = (const float*)d_in[2];
  const float* WQ    = (const float*)d_in[3];
  const float* WK    = (const float*)d_in[4];
  const float* WV    = (const float*)d_in[5];

  unsigned short* Qb = (unsigned short*)d_ws;                 // 16 MB
  unsigned short* Kb = Qb + (size_t)MROWS * OUT_DIM;          // 16 MB
  unsigned short* Vb = Kb + (size_t)MROWS * OUT_DIM;          // 16 MB

  dim3 gProj(OUT_DIM / 64, MROWS / 128);   // (16, 64)
  dim3 blk(256);
  proj_gemm_bf16<<<gProj, blk, 0, stream>>>(query, WQ, Qb, 0, 0.125f);  // 1/sqrt(64)
  proj_gemm_bf16<<<gProj, blk, 0, stream>>>(key,   WK, Kb, 0, 1.0f);
  proj_gemm_bf16<<<gProj, blk, 0, stream>>>(value, WV, Vb, 1, 1.0f);

  dim3 gAttn(SEQ / 128, BATCH * HEADS);    // (16, 64)
  flash_attn<<<gAttn, blk, 0, stream>>>(Qb, Kb, Vb, (float*)d_out);
}
